// attention_22874995818902
// MI455X (gfx1250) — matrix operand order repaired and checked by reading, not compiled, not run on hardware
//
#include <hip/hip_runtime.h>
#include <hip/hip_bf16.h>

// ---------------------------------------------------------------------------
// Attention fwd:  q = query·Wqᵀ+bq ; k = key·Wkᵀ+bk ; v = key·Wvᵀ+bv
//                 S = mask(q·kᵀ/32) ; P = softmax(S) (NaN->0) ; out = P·v
// All GEMMs: bf16 WMMA (v_wmma_f32_16x16x32_bf16), f32 accumulate.
// Scores/PV GEMMs stage tiles via the Tensor Data Mover (double-buffered LDS).
// ---------------------------------------------------------------------------

#define B_   8
#define SQ_  2048
#define SK_  2048
#define HID_ 1024

#if defined(__HIP_DEVICE_COMPILE__) && \
    __has_builtin(__builtin_amdgcn_tensor_load_to_lds) && \
    __has_builtin(__builtin_amdgcn_s_wait_tensorcnt)
#define HAS_TDM 1
#else
#define HAS_TDM 0
#endif

typedef __attribute__((ext_vector_type(16))) __bf16       v16bf;
typedef __attribute__((ext_vector_type(8)))  float        v8f;
typedef __attribute__((ext_vector_type(4)))  unsigned int v4u_t;
typedef __attribute__((ext_vector_type(8)))  int          v8i_t;
typedef __attribute__((ext_vector_type(4)))  int          v4i_t;

union Frag { uint4 u[2]; v16bf v; };

__device__ __forceinline__ unsigned short f2bf(float f) {
    unsigned int u = __float_as_uint(f);
    u += 0x7FFFu + ((u >> 16) & 1u);          // round-to-nearest-even
    return (unsigned short)(u >> 16);
}
__device__ __forceinline__ unsigned int pk2(float a, float b) {
    return (unsigned int)f2bf(a) | ((unsigned int)f2bf(b) << 16);
}
__device__ __forceinline__ uint4 pack8(float4 f0, float4 f1) {
    uint4 u;
    u.x = pk2(f0.x, f0.y); u.y = pk2(f0.z, f0.w);
    u.z = pk2(f1.x, f1.y); u.w = pk2(f1.z, f1.w);
    return u;
}

#if HAS_TDM
// TDM: copy one 128-row x 32-col bf16 tile (row stride ldK elements) from
// global into LDS laid out with 80-byte padded rows (64B data + 16B pad).
__device__ __forceinline__ void tdm_load_tile_128x32(const unsigned short* gtile,
                                                     unsigned lds_byte, int ldK)
{
    unsigned long long ga = (unsigned long long)gtile;
    v4u_t g0;
    g0.x = 1u;                                           // count=1, user mode
    g0.y = lds_byte;                                     // LDS byte address
    g0.z = (unsigned)ga;                                 // global_addr[31:0]
    g0.w = (unsigned)((ga >> 32) & 0x1FFFFFFull)         // global_addr[56:32]
         | (2u << 30);                                   // type = 2 ("image")
    v8i_t g1;
    g1[0] = (int)((1u << 16)      // data_size = 2 bytes
                | (1u << 20)      // pad_enable
                | (3u << 22)      // pad_interval: every 16 DWORDs (64B row)
                | (3u << 25));    // pad_amount: 4 DWORDs (16B)
    g1[1] = (int)(32u  << 16);    // tensor_dim0 = 32 (low 16 bits)
    g1[2] = (int)(128u << 16);    // tensor_dim0 hi | tensor_dim1 = 128 (lo)
    g1[3] = (int)(32u  << 16);    // tensor_dim1 hi | tile_dim0 = 32
    g1[4] = 128;                  // tile_dim1 = 128, tile_dim2 = 0
    g1[5] = ldK;                  // tensor_dim0_stride[31:0] (elements)
    g1[6] = 0;                    // stride hi | tensor_dim1_stride lo
    g1[7] = 0;
    v4i_t z4 = {0, 0, 0, 0};
    v8i_t z8 = {0, 0, 0, 0, 0, 0, 0, 0};
    __builtin_amdgcn_tensor_load_to_lds(g0, g1, z4, z4, z8, 0);
}
#endif

// MODE 0: C_bf16[m*N+n] = acc + bias[n]
// MODE 1: C_bf16[(m/rpb)*cBS + n*rpb + (m%rpb)] = acc + bias[n]  (transposed)
// MODE 2: C_f32[b*cBS + m*N + n] = mask ? acc*scale : -inf
// MODE 3: C_f32[b*cBS + m*N + n] = acc
template<int MODE, bool A32, bool B32, bool ASYNC>
__global__ __launch_bounds__(256)
void gemm_bf16_wmma(const void* __restrict__ Ap, const void* __restrict__ Bp,
                    void* __restrict__ Cp, const float* __restrict__ bias,
                    const int* __restrict__ maskp,
                    int M, int N, int K,
                    long long aBS, long long bBS, long long cBS,
                    float scale, int rpb)
{
    constexpr int NBUF = ASYNC ? 2 : 1;
    __shared__ __align__(16) unsigned short lsA[NBUF][128][40];  // 32 K + 8 pad
    __shared__ __align__(16) unsigned short lsB[NBUF][128][40];

    const int tid    = threadIdx.x;
    const int lane   = tid & 31;
    const int wave   = tid >> 5;
    const int wm     = wave & 3;          // 4 waves along M (32 rows each)
    const int wn     = wave >> 2;         // 2 waves along N (64 cols each)
    const int batch  = blockIdx.z;
    const int blockM = blockIdx.y * 128;
    const int blockN = blockIdx.x * 128;

    const float*          Af = (const float*)Ap;
    const unsigned short* Ab = (const unsigned short*)Ap;
    const float*          Bf = (const float*)Bp;
    const unsigned short* Bb = (const unsigned short*)Bp;

    v8f acc[2][4];
    const v8f vzero = {0.f,0.f,0.f,0.f,0.f,0.f,0.f,0.f};
#pragma unroll
    for (int mt = 0; mt < 2; ++mt)
#pragma unroll
        for (int nt = 0; nt < 4; ++nt) acc[mt][nt] = vzero;

    const int rl  = lane & 15;
    const int klo = (lane < 16) ? 0 : 8;        // A-operand K-chunk select
    const int kb0 = (lane >> 4) << 4;           // B-operand K-half select
    const int nsteps = K / 32;

#if HAS_TDM
    if constexpr (ASYNC) {                      // prefetch first tile pair
        if (wave == 0) {
            tdm_load_tile_128x32(Ab + (size_t)batch * aBS + (size_t)blockM * K,
                                 (unsigned)(unsigned long long)&lsA[0][0][0], K);
            tdm_load_tile_128x32(Bb + (size_t)batch * bBS + (size_t)blockN * K,
                                 (unsigned)(unsigned long long)&lsB[0][0][0], K);
        }
    }
#endif

    for (int it = 0; it < nsteps; ++it) {
        const int kk = it * 32;
        int  bi = 0;
        bool staged = false;
#if HAS_TDM
        if constexpr (ASYNC) {
            if (wave == 0) __builtin_amdgcn_s_wait_tensorcnt(0);
            __syncthreads();                    // tile visible; prev buf free
            bi = it & 1;
            if (wave == 0 && it + 1 < nsteps) { // DMA next tile during compute
                int nk = kk + 32;
                tdm_load_tile_128x32(
                    Ab + (size_t)batch * aBS + (size_t)blockM * K + nk,
                    (unsigned)(unsigned long long)&lsA[bi ^ 1][0][0], K);
                tdm_load_tile_128x32(
                    Bb + (size_t)batch * bBS + (size_t)blockN * K + nk,
                    (unsigned)(unsigned long long)&lsB[bi ^ 1][0][0], K);
            }
            staged = true;
        }
#endif
        if (!staged) {
            __syncthreads();   // previous iteration's compute finished
#pragma unroll
            for (int i = 0; i < 2; ++i) {
                int c  = tid + i * 256;         // 0..511 16B chunks
                int r  = c >> 2;
                int cc = (c & 3) << 3;          // 0,8,16,24
                size_t aoff = (size_t)batch * aBS + (size_t)(blockM + r) * K + kk + cc;
                if (A32) {
                    float4 f0 = *(const float4*)(Af + aoff);
                    float4 f1 = *(const float4*)(Af + aoff + 4);
                    *(uint4*)&lsA[0][r][cc] = pack8(f0, f1);
                } else {
                    *(uint4*)&lsA[0][r][cc] = *(const uint4*)(Ab + aoff);
                }
                size_t boff = (size_t)batch * bBS + (size_t)(blockN + r) * K + kk + cc;
                if (B32) {
                    float4 f0 = *(const float4*)(Bf + boff);
                    float4 f1 = *(const float4*)(Bf + boff + 4);
                    *(uint4*)&lsB[0][r][cc] = pack8(f0, f1);
                } else {
                    *(uint4*)&lsB[0][r][cc] = *(const uint4*)(Bb + boff);
                }
            }
            __syncthreads();
        }

        // ---- gather WMMA fragments (CDNA5 wave32 layouts) ----
        Frag fa[2], fb[4];
#pragma unroll
        for (int mt = 0; mt < 2; ++mt) {
            int r = wm * 32 + mt * 16 + rl;     // M = lane%16
            fa[mt].u[0] = *(const uint4*)&lsA[bi][r][klo];       // K 0..7 / 8..15
            fa[mt].u[1] = *(const uint4*)&lsA[bi][r][klo + 16];  // K 16..23/24..31
        }
#pragma unroll
        for (int nt = 0; nt < 4; ++nt) {
            int n = wn * 64 + nt * 16 + rl;     // N = lane%16
            fb[nt].u[0] = *(const uint4*)&lsB[bi][n][kb0 >> 1];        // K 0..7 / 8..15
            fb[nt].u[1] = *(const uint4*)&lsB[bi][n][(kb0 >> 1) + 16]; // K 16..23/24..31
        }
#pragma unroll
        for (int mt = 0; mt < 2; ++mt)
#pragma unroll
            for (int nt = 0; nt < 4; ++nt)
                acc[mt][nt] = __builtin_amdgcn_wmma_f32_16x16x32_bf16(
                    false, fa[mt].v, false, fb[nt].v,
                    (short)0, acc[mt][nt], false, false);
    }

    // ---- epilogue ----
    const int mhi = (lane >> 4) << 3;           // lanes 16-31 hold M+8
#pragma unroll
    for (int mt = 0; mt < 2; ++mt) {
#pragma unroll
        for (int nt = 0; nt < 4; ++nt) {
            int n = blockN + wn * 64 + nt * 16 + rl;
            float bsv = (MODE <= 1) ? bias[n] : 0.0f;
#pragma unroll
            for (int g = 0; g < 8; ++g) {
                int   m = blockM + wm * 32 + mt * 16 + mhi + g;
                float v = acc[mt][nt][g];
                if (MODE == 0) {
                    ((unsigned short*)Cp)[(size_t)m * N + n] = f2bf(v + bsv);
                } else if (MODE == 1) {
                    int bb = m / rpb, s = m % rpb;
                    ((unsigned short*)Cp)[(size_t)bb * cBS + (size_t)n * rpb + s] =
                        f2bf(v + bsv);
                } else if (MODE == 2) {
                    size_t idx = (size_t)batch * cBS + (size_t)m * N + n;
                    float sv = v * scale;
                    if (maskp[idx] == 0) sv = -__builtin_inff();
                    ((float*)Cp)[idx] = sv;
                } else {
                    ((float*)Cp)[(size_t)batch * cBS + (size_t)m * N + n] = v;
                }
            }
        }
    }
}

// Row softmax over SK=2048 with fully-masked-row -> 0 cleanup.
// Writes normalized fp32 back in place and a bf16 copy for the PV GEMM.
__global__ __launch_bounds__(256)
void softmax_kernel(float* __restrict__ score, unsigned short* __restrict__ P)
{
    const int row = blockIdx.x;
    const int tid = threadIdx.x;
    float*          s = score + (size_t)row * SK_;
    unsigned short* p = P     + (size_t)row * SK_;

    __shared__ float red[256];

    float x[8];
    float mx = -__builtin_inff();
#pragma unroll
    for (int j = 0; j < 8; ++j) { x[j] = s[tid + j * 256]; mx = fmaxf(mx, x[j]); }
    red[tid] = mx; __syncthreads();
    for (int off = 128; off > 0; off >>= 1) {
        if (tid < off) red[tid] = fmaxf(red[tid], red[tid + off]);
        __syncthreads();
    }
    mx = red[0]; __syncthreads();

    const bool dead = (mx == -__builtin_inff());   // fully masked row
    float e[8], sum = 0.f;
#pragma unroll
    for (int j = 0; j < 8; ++j) {
        e[j] = dead ? 0.0f : exp2f((x[j] - mx) * 1.44269504f);
        sum += e[j];
    }
    red[tid] = sum; __syncthreads();
    for (int off = 128; off > 0; off >>= 1) {
        if (tid < off) red[tid] += red[tid + off];
        __syncthreads();
    }
    sum = red[0];
    const float inv = (sum > 0.f) ? 1.0f / sum : 0.0f;
#pragma unroll
    for (int j = 0; j < 8; ++j) {
        float pv = e[j] * inv;
        s[tid + j * 256] = pv;
        p[tid + j * 256] = f2bf(pv);
    }
}

extern "C" void kernel_launch(void* const* d_in, const int* in_sizes, int n_in,
                              void* d_out, int out_size, void* d_ws, size_t ws_size,
                              hipStream_t stream) {
    const float* key   = (const float*)d_in[0];
    const float* query = (const float*)d_in[1];
    const int*   mask  = (const int*)d_in[2];
    const float* Wq    = (const float*)d_in[3];
    const float* bq    = (const float*)d_in[4];
    const float* Wk    = (const float*)d_in[5];
    const float* bk    = (const float*)d_in[6];
    const float* Wv    = (const float*)d_in[7];
    const float* bv    = (const float*)d_in[8];

    float* out       = (float*)d_out;                          // [B,SQ,HID]
    float* score_out = out + (size_t)B_ * SQ_ * HID_;          // [B,SQ,SK]

    // workspace: q | k | vT | P  (all bf16)
    unsigned short* qb = (unsigned short*)d_ws;
    unsigned short* kb = qb + (size_t)B_ * SQ_ * HID_;
    unsigned short* vT = kb + (size_t)B_ * SK_ * HID_;
    unsigned short* Pb = vT + (size_t)B_ * SK_ * HID_;

    const dim3 blk(256);
    const long long qkS = (long long)SQ_ * HID_;   // per-batch q/k/v stride
    const long long ssS = (long long)SQ_ * SK_;    // per-batch score stride

    // QKV projections: M = B*SQ = 16384, N = HID, K = 1024 (fp32 in, bf16 out)
    gemm_bf16_wmma<0, true, true, false><<<dim3(HID_/128, (B_*SQ_)/128, 1), blk, 0, stream>>>(
        query, Wq, qb, bq, nullptr, B_*SQ_, HID_, HID_, 0, 0, 0, 1.0f, 0);
    gemm_bf16_wmma<0, true, true, false><<<dim3(HID_/128, (B_*SK_)/128, 1), blk, 0, stream>>>(
        key, Wk, kb, bk, nullptr, B_*SK_, HID_, HID_, 0, 0, 0, 1.0f, 0);
    gemm_bf16_wmma<1, true, true, false><<<dim3(HID_/128, (B_*SK_)/128, 1), blk, 0, stream>>>(
        key, Wv, vT, bv, nullptr, B_*SK_, HID_, HID_,
        0, 0, (long long)HID_ * SK_, 1.0f, SK_);

    // scores: per batch M=SQ, N=SK, K=HID ; scale=1/sqrt(HID)=1/32 ; mask -> -inf
    // (TDM double-buffered staging)
    gemm_bf16_wmma<2, false, false, true><<<dim3(SK_/128, SQ_/128, B_), blk, 0, stream>>>(
        qb, kb, score_out, nullptr, mask, SQ_, SK_, HID_,
        qkS, qkS, ssS, 1.0f / 32.0f, 0);

    // softmax over SK per row (also emits bf16 P)
    softmax_kernel<<<dim3(B_ * SQ_), blk, 0, stream>>>(score_out, Pb);

    // out = P · v : per batch M=SQ, N=HID, K=SK (B operand = vT, K-contiguous)
    // (TDM double-buffered staging)
    gemm_bf16_wmma<3, false, false, true><<<dim3(HID_/128, SQ_/128, B_), blk, 0, stream>>>(
        Pb, vT, out, nullptr, nullptr, SQ_, HID_, SK_,
        ssS, (long long)HID_ * SK_, qkS, 1.0f, 0);
}
